// DotProductAttention_1683627180789
// MI455X (gfx1250) — matrix operand order repaired and checked by reading, not compiled, not run on hardware
//
#include <hip/hip_runtime.h>

// CDNA5 / gfx1250 flash-attention, bf16 WMMA with f32 accumulate.
//
// Pass 1: convert K and V f32 -> bf16 once, stored pre-swizzled in the exact
// WMMA A-fragment per-lane layout (workspace). Inner loop fragment loads are
// then plain 32B (2 x global_load_b128) per fragment.
//
// Pass 2: one wave32 per 16-query tile; keys streamed in 32-key chunks with
// online softmax. S^T = K*Q^T puts the softmax key-axis on vector elements
// (+ one lane^16 exchange), and P^T lands in the WMMA B-fragment layout for
// O^T += V^T * P^T. Cross-half traffic minimized: no swap needed for
// P^T, 1 shuffle for the chunk max, and the softmax denominator is kept as a
// per-half partial sum combined once after the loop. Chunks entirely past
// valid_len are skipped (bit-exact vs reference: exp(-1e6 - m) underflows to
// 0.0f there too); only the straddling chunk runs the masked path.

typedef __attribute__((ext_vector_type(16))) __bf16 v16bf;
typedef __attribute__((ext_vector_type(8)))  float  v8f;

#define ATT_B  16
#define ATT_SQ 2048
#define ATT_SK 2048
#define ATT_D  64
#define NCH    (ATT_SK / 32)              // 64 key-chunks
#define CHUNK_ELEMS (4 * 32 * 16)         // 4 fragments x 32 lanes x 16 bf16

__device__ __forceinline__ v8f wmma_bf16(v16bf a, v16bf b, v8f c) {
  return __builtin_amdgcn_wmma_f32_16x16x32_bf16(
      false, a, false, b, (short)0, c, false, false);
}

// ---------------------------------------------------------------------------
// Pass 1a: K[b][key][d] (f32) -> wsK, A-fragment layout.
// Fragment (chunk ch, key-tile t2, d-slab f): lane l holds key = 32ch+16t2+(l&15),
// element e -> d = 32f + 8*(l>>4) + (e&7) + 16*(e>=8).
__global__ __launch_bounds__(256) void swizzle_k_kernel(
    const float* __restrict__ Kg, unsigned short* __restrict__ wsK_us)
{
  const int tid = blockIdx.x * 256 + threadIdx.x;
  const int l  = tid & 31;
  const int f  = (tid >> 5) & 1;
  const int t2 = (tid >> 6) & 1;
  const int ch = (tid >> 7) & (NCH - 1);
  const int b  = tid >> 13;
  const int key = ch * 32 + t2 * 16 + (l & 15);
  const float* src = Kg + ((size_t)b * ATT_SK + key) * ATT_D + f * 32 + (l >> 4) * 8;
  __bf16* dst = (__bf16*)wsK_us + (size_t)tid * 16;
#pragma unroll
  for (int e = 0; e < 16; ++e)
    dst[e] = (__bf16)src[(e & 7) + ((e >> 3) << 4)];
}

// Pass 1b: V[b][key][d] (f32) -> wsV, V^T A-fragment layout.
// Fragment (chunk ch, d-tile t): lane l holds d = 16t+(l&15),
// element e -> key = 32ch + 8*(l>>4) + (e&7) + 16*(e>=8).
__global__ __launch_bounds__(256) void swizzle_v_kernel(
    const float* __restrict__ Vg, unsigned short* __restrict__ wsV_us)
{
  const int tid = blockIdx.x * 256 + threadIdx.x;
  const int l  = tid & 31;
  const int t  = (tid >> 5) & 3;
  const int ch = (tid >> 7) & (NCH - 1);
  const int b  = tid >> 13;
  const float* src = Vg + ((size_t)b * ATT_SK + ch * 32 + (l >> 4) * 8) * ATT_D
                   + t * 16 + (l & 15);
  __bf16* dst = (__bf16*)wsV_us + (size_t)tid * 16;
#pragma unroll
  for (int e = 0; e < 16; ++e)
    dst[e] = (__bf16)src[(size_t)((e & 7) + ((e >> 3) << 4)) * ATT_D];
}

// ---------------------------------------------------------------------------
// Online-softmax update for one 32-key chunk of scores (S^T in st0/st1).
// Produces the P^T B-fragment. s_run is a PER-HALF partial sum (combined once
// after the key loop); m_run is shared across halves.
template <bool MASKED>
__device__ __forceinline__ v16bf softmax_update(
    v8f st0, v8f st1, int c, int vlen, int lh,
    v8f& o0, v8f& o1, v8f& o2, v8f& o3,
    float& m_run, float& s_run)
{
  float p0[8], p1[8];
  float lm = -INFINITY;
#pragma unroll
  for (int v = 0; v < 8; ++v) {
    float x0 = st0[v];
    float x1 = st1[v];
    if (MASKED) {
      const int k0 = c + lh * 8 + v;
      x0 = (k0      < vlen) ? x0 : -1.0e6f;
      x1 = (k0 + 16 < vlen) ? x1 : -1.0e6f;
    }
    p0[v] = x0; p1[v] = x1;
    lm = fmaxf(lm, fmaxf(x0, x1));
  }
  lm = fmaxf(lm, __shfl_xor(lm, 16, 32));            // chunk max, shared
  const float m_new = fmaxf(m_run, lm);
  const float alpha = __expf(m_run - m_new);         // first chunk: exp(-inf)=0
  float ls = 0.0f;
#pragma unroll
  for (int v = 0; v < 8; ++v) {
    p0[v] = __expf(p0[v] - m_new);
    p1[v] = __expf(p1[v] - m_new);
    ls += p0[v] + p1[v];
  }
  s_run = s_run * alpha + ls;                        // per-half partial
  m_run = m_new;
#pragma unroll
  for (int v = 0; v < 8; ++v) {
    o0[v] *= alpha; o1[v] *= alpha; o2[v] *= alpha; o3[v] *= alpha;
  }

  // P^T -> B-fragment. Pack bf16 pairs into dwords, no swap is needed: each
  // lane half already holds its keys 8lh+v (pk0) and 16+8lh+v (pk1).
  unsigned pk0[4], pk1[4];
#pragma unroll
  for (int w = 0; w < 4; ++w) {
    union { unsigned u; __bf16 h[2]; } c0, c1;
    c0.h[0] = (__bf16)p0[2 * w]; c0.h[1] = (__bf16)p0[2 * w + 1];
    c1.h[0] = (__bf16)p1[2 * w]; c1.h[1] = (__bf16)p1[2 * w + 1];
    pk0[w] = c0.u; pk1[w] = c1.u;
  }
  union { v16bf v; unsigned u[8]; } pbu;
#pragma unroll
  for (int w = 0; w < 4; ++w) {
    const unsigned send = (lh == 0) ? pk1[w] : pk0[w];
    const unsigned recv = (unsigned)__shfl_xor((int)send, 16, 32);
    pbu.u[w]     = pk0[w];                           // elems 2w,2w+1
    pbu.u[w + 4] = pk1[w];                           // elems 2w+8,2w+9
  }
  return pbu.v;
}

// One 32-key chunk from the pre-swizzled bf16 streams.
template <bool MASKED>
__device__ __forceinline__ void process_chunk(
    int c, int vlen, int lh, int lane,
    const unsigned short* __restrict__ kc, const unsigned short* __restrict__ vc,
    const v16bf (&qb)[2],
    v8f& o0, v8f& o1, v8f& o2, v8f& o3,
    float& m_run, float& s_run)
{
  const int loff = lane * 16;
  const __bf16* kb = (const __bf16*)kc;
  const __bf16* vb = (const __bf16*)vc;

  const v16bf ka00 = *(const v16bf*)(kb + 0 * 512 + loff);  // keys c..c+15,  d 0..31
  const v16bf ka01 = *(const v16bf*)(kb + 1 * 512 + loff);  // keys c..c+15,  d 32..63
  const v16bf ka10 = *(const v16bf*)(kb + 2 * 512 + loff);  // keys c+16..31, d 0..31
  const v16bf ka11 = *(const v16bf*)(kb + 3 * 512 + loff);  // keys c+16..31, d 32..63

  v8f st0 = {}, st1 = {};
  st0 = wmma_bf16(ka00, qb[0], st0);
  st0 = wmma_bf16(ka01, qb[1], st0);
  st1 = wmma_bf16(ka10, qb[0], st1);
  st1 = wmma_bf16(ka11, qb[1], st1);

  const v16bf va0 = *(const v16bf*)(vb + 0 * 512 + loff);
  const v16bf va1 = *(const v16bf*)(vb + 1 * 512 + loff);
  const v16bf va2 = *(const v16bf*)(vb + 2 * 512 + loff);
  const v16bf va3 = *(const v16bf*)(vb + 3 * 512 + loff);

  const v16bf pb = softmax_update<MASKED>(st0, st1, c, vlen, lh,
                                          o0, o1, o2, o3, m_run, s_run);

  o0 = wmma_bf16(va0, pb, o0);
  o1 = wmma_bf16(va1, pb, o1);
  o2 = wmma_bf16(va2, pb, o2);
  o3 = wmma_bf16(va3, pb, o3);
}

__global__ __launch_bounds__(256, 1) void flash_attn_pre(
    const float* __restrict__ Qg, const int* __restrict__ VLg,
    const unsigned short* __restrict__ wsK, const unsigned short* __restrict__ wsV,
    float* __restrict__ Og)
{
  const int lane = threadIdx.x & 31;
  const int wave = threadIdx.x >> 5;
  const int tile = blockIdx.x * 8 + wave;       // [0, 2048)
  const int b     = tile >> 7;
  const int qbase = (tile & 127) << 4;
  const int vlen  = VLg[b];
  const int lh = lane >> 4;
  const int ln = lane & 15;

  // Q^T B-fragments (1/sqrt(64) folded): lane -> q = ln, elem e -> d = 32f+8lh+(e&7)+16*(e>=8)
  const float* Qrow = Qg + ((size_t)b * ATT_SQ + qbase + ln) * ATT_D;
  v16bf qb[2];
#pragma unroll
  for (int f = 0; f < 2; ++f)
#pragma unroll
    for (int e = 0; e < 16; ++e)
      qb[f][e] = (__bf16)(Qrow[f * 32 + lh * 8 + (e & 7) + ((e >> 3) << 4)] * 0.125f);

  v8f o0 = {}, o1 = {}, o2 = {}, o3 = {};
  float m_run = -INFINITY;
  float s_run = 0.0f;                            // per-half partial sum

  const unsigned short* kbase = wsK + (size_t)b * NCH * CHUNK_ELEMS;
  const unsigned short* vbase = wsV + (size_t)b * NCH * CHUNK_ELEMS;

  const int full = vlen >> 5;                    // chunks fully below valid_len
  for (int ch = 0; ch < full; ++ch) {
    const int chn = (ch + 1 < NCH) ? ch + 1 : ch;
    __builtin_prefetch(kbase + (size_t)chn * CHUNK_ELEMS + lane * 16, 0, 3);
    __builtin_prefetch(vbase + (size_t)chn * CHUNK_ELEMS + lane * 16, 0, 3);
    process_chunk<false>(ch * 32, vlen, lh, lane,
                         kbase + (size_t)ch * CHUNK_ELEMS,
                         vbase + (size_t)ch * CHUNK_ELEMS,
                         qb, o0, o1, o2, o3, m_run, s_run);
  }
  if (vlen & 31) {
    process_chunk<true>(full * 32, vlen, lh, lane,
                        kbase + (size_t)full * CHUNK_ELEMS,
                        vbase + (size_t)full * CHUNK_ELEMS,
                        qb, o0, o1, o2, o3, m_run, s_run);
  }
  // chunks with c >= vlen contribute exactly 0 (exp underflow) -> skipped.

  const float s_tot = s_run + __shfl_xor(s_run, 16, 32);  // combine halves once
  const float inv = 1.0f / s_tot;
  float* orow = Og + ((size_t)b * ATT_SQ + qbase + ln) * ATT_D;
#pragma unroll
  for (int v = 0; v < 8; ++v) {
    const int d = lh * 8 + v;
    orow[d]      = o0[v] * inv;
    orow[d + 16] = o1[v] * inv;
    orow[d + 32] = o2[v] * inv;
    orow[d + 48] = o3[v] * inv;
  }
}

// ---------------------------------------------------------------------------
// Fallback (workspace too small): single pass, gathers + converts K/V from
// f32 on the fly; shares the softmax/P^T helper.
__global__ __launch_bounds__(256, 1) void flash_attn_fallback(
    const float* __restrict__ Qg, const float* __restrict__ Kg,
    const float* __restrict__ Vg, const int* __restrict__ VLg,
    float* __restrict__ Og)
{
  const int lane = threadIdx.x & 31;
  const int wave = threadIdx.x >> 5;
  const int tile = blockIdx.x * 8 + wave;
  const int b     = tile >> 7;
  const int qbase = (tile & 127) << 4;
  const int vlen  = VLg[b];
  const int lh = lane >> 4;
  const int ln = lane & 15;

  const float* Qrow = Qg + ((size_t)b * ATT_SQ + qbase + ln) * ATT_D;
  const float* Kb   = Kg + (size_t)b * ATT_SK * ATT_D;
  const float* Vb   = Vg + (size_t)b * ATT_SK * ATT_D;

  v16bf qb[2];
#pragma unroll
  for (int f = 0; f < 2; ++f)
#pragma unroll
    for (int e = 0; e < 16; ++e)
      qb[f][e] = (__bf16)(Qrow[f * 32 + lh * 8 + (e & 7) + ((e >> 3) << 4)] * 0.125f);

  v8f o0 = {}, o1 = {}, o2 = {}, o3 = {};
  float m_run = -INFINITY, s_run = 0.0f;

  for (int c = 0; c < ATT_SK; c += 32) {
    if (c >= vlen) break;
    v16bf ka[2][2];
#pragma unroll
    for (int t2 = 0; t2 < 2; ++t2) {
      const float* krow = Kb + (size_t)(c + t2 * 16 + ln) * ATT_D + lh * 8;
#pragma unroll
      for (int f = 0; f < 2; ++f)
#pragma unroll
        for (int e = 0; e < 16; ++e)
          ka[t2][f][e] = (__bf16)krow[f * 32 + (e & 7) + ((e >> 3) << 4)];
    }
    v8f st0 = {}, st1 = {};
    st0 = wmma_bf16(ka[0][0], qb[0], st0);
    st0 = wmma_bf16(ka[0][1], qb[1], st0);
    st1 = wmma_bf16(ka[1][0], qb[0], st1);
    st1 = wmma_bf16(ka[1][1], qb[1], st1);

    v16bf va[4];
#pragma unroll
    for (int t = 0; t < 4; ++t) {
      const float* vcol = Vb + (size_t)c * ATT_D + t * 16 + ln;
#pragma unroll
      for (int e = 0; e < 16; ++e) {
        const int koff = lh * 8 + (e & 7) + ((e >> 3) << 4);
        va[t][e] = (__bf16)vcol[(size_t)koff * ATT_D];
      }
    }

    const v16bf pb = softmax_update<true>(st0, st1, c, vlen, lh,
                                          o0, o1, o2, o3, m_run, s_run);
    o0 = wmma_bf16(va[0], pb, o0);
    o1 = wmma_bf16(va[1], pb, o1);
    o2 = wmma_bf16(va[2], pb, o2);
    o3 = wmma_bf16(va[3], pb, o3);
  }

  const float s_tot = s_run + __shfl_xor(s_run, 16, 32);
  const float inv = 1.0f / s_tot;
  float* orow = Og + ((size_t)b * ATT_SQ + qbase + ln) * ATT_D;
#pragma unroll
  for (int v = 0; v < 8; ++v) {
    const int d = lh * 8 + v;
    orow[d]      = o0[v] * inv;
    orow[d + 16] = o1[v] * inv;
    orow[d + 32] = o2[v] * inv;
    orow[d + 48] = o3[v] * inv;
  }
}

// ---------------------------------------------------------------------------
extern "C" void kernel_launch(void* const* d_in, const int* in_sizes, int n_in,
                              void* d_out, int out_size, void* d_ws, size_t ws_size,
                              hipStream_t stream) {
  (void)in_sizes; (void)n_in; (void)out_size;
  const float* Q  = (const float*)d_in[0];
  const float* K  = (const float*)d_in[1];
  const float* V  = (const float*)d_in[2];
  const int*   VL = (const int*)d_in[3];
  float* O = (float*)d_out;

  const size_t elems = (size_t)ATT_B * ATT_SK * ATT_D;     // 2,097,152
  const size_t need  = 2 * elems * sizeof(unsigned short); // 8 MiB

  if (d_ws != nullptr && ws_size >= need) {
    unsigned short* wsK = (unsigned short*)d_ws;
    unsigned short* wsV = wsK + elems;
    const int conv_blocks = (int)(elems / 16 / 256);       // 512
    hipLaunchKernelGGL(swizzle_k_kernel, dim3(conv_blocks), dim3(256), 0, stream, K, wsK);
    hipLaunchKernelGGL(swizzle_v_kernel, dim3(conv_blocks), dim3(256), 0, stream, V, wsV);
    hipLaunchKernelGGL(flash_attn_pre, dim3(256), dim3(256), 0, stream, Q, VL, wsK, wsV, O);
  } else {
    hipLaunchKernelGGL(flash_attn_fallback, dim3(256), dim3(256), 0, stream, Q, K, V, VL, O);
  }
}
